// KroneckerMixer_13443247637017
// MI455X (gfx1250) — compile-verified, matrix operand order checked by reading, not run on hardware
//
#include <hip/hip_runtime.h>
#include <hip/hip_bf16.h>

// ---------- problem constants ----------
#define BB   256
#define NN   1024
#define KK   64
#define NK   65536        // NN*KK
#define TEMP_INV 5.0f     // 1/0.2
#define LN_EPS   1e-5f

typedef __attribute__((ext_vector_type(16))) _Float16 v16h;
typedef __attribute__((ext_vector_type(8)))  _Float16 v8h;
typedef __attribute__((ext_vector_type(8)))  float    v8f;

union V8F { v8f v; float f[8]; };

// ---------- CDNA5 async global->LDS support (guarded; falls back cleanly) ----------
#if defined(__HIP_DEVICE_COMPILE__) && __has_builtin(__builtin_amdgcn_global_load_async_to_lds_b128)
#define HAVE_ASYNC_LDS 1
#else
#define HAVE_ASYNC_LDS 0
#endif

#if HAVE_ASYNC_LDS
typedef int v4i __attribute__((vector_size(16)));     // matches builtin param type
typedef __attribute__((address_space(1))) v4i gv4i;   // global (prints as __device__)
typedef __attribute__((address_space(3))) v4i lv4i;   // LDS    (prints as __shared__)
static __device__ __forceinline__ gv4i* asGlobal(const void* p) {
  return (gv4i*)(unsigned long long)p;                // AS1 is 64-bit
}
static __device__ __forceinline__ lv4i* asLds(void* p) {
  return (lv4i*)(unsigned)(unsigned long long)p;      // low 32 bits = LDS offset
}
#if __has_builtin(__builtin_amdgcn_s_wait_asynccnt)
#define WAIT_ASYNC() __builtin_amdgcn_s_wait_asynccnt(0)
#else
#define WAIT_ASYNC() asm volatile("s_wait_asynccnt 0" ::: "memory")
#endif
#endif

// ---------- block reductions (256 threads) ----------
static __device__ __forceinline__ float blockMax(float v, float* red, int tid) {
  red[tid] = v; __syncthreads();
  #pragma unroll
  for (int s = 128; s > 0; s >>= 1) {
    if (tid < s) red[tid] = fmaxf(red[tid], red[tid + s]);
    __syncthreads();
  }
  float r = red[0]; __syncthreads(); return r;
}
static __device__ __forceinline__ float blockSum(float v, float* red, int tid) {
  red[tid] = v; __syncthreads();
  #pragma unroll
  for (int s = 128; s > 0; s >>= 1) {
    if (tid < s) red[tid] += red[tid + s];
    __syncthreads();
  }
  float r = red[0]; __syncthreads(); return r;
}

// assemble a 16-half WMMA fragment from two contiguous 8-half LDS runs
static __device__ __forceinline__ v16h ldsFrag(const _Float16* base) {
  v8h lo = *(const v8h*)(base);
  v8h hi = *(const v8h*)(base + 16);
  return __builtin_shufflevector(lo, hi, 0,1,2,3,4,5,6,7,8,9,10,11,12,13,14,15);
}

// ---------- kernel 1: Sinkhorn on A, row pass -> la1 = A/T - rowLSE ----------
__global__ __launch_bounds__(256) void kSinkARow(const float* __restrict__ Alog,
                                                 float* __restrict__ la1) {
  __shared__ float red[256];
  const int i = blockIdx.x, tid = threadIdx.x;
  const float* row = Alog + (size_t)i * NN;
  float mx = -1e30f;
  for (int j = tid; j < NN; j += 256) mx = fmaxf(mx, row[j] * TEMP_INV);
  mx = blockMax(mx, red, tid);
  float s = 0.f;
  for (int j = tid; j < NN; j += 256) s += __expf(row[j] * TEMP_INV - mx);
  s = blockSum(s, red, tid);
  const float lse = mx + __logf(s);
  float* orow = la1 + (size_t)i * NN;
  for (int j = tid; j < NN; j += 256) orow[j] = row[j] * TEMP_INV - lse;
}

// ---------- kernel 2: Sinkhorn on A, col pass -> At[m][n] = A[n][m] (f16) ----------
__global__ __launch_bounds__(256) void kSinkACol(const float* __restrict__ la1,
                                                 _Float16* __restrict__ At) {
  __shared__ float red[256];
  const int j = blockIdx.x, tid = threadIdx.x;
  float mx = -1e30f;
  for (int i = tid; i < NN; i += 256) mx = fmaxf(mx, la1[(size_t)i * NN + j]);
  mx = blockMax(mx, red, tid);
  float s = 0.f;
  for (int i = tid; i < NN; i += 256) s += __expf(la1[(size_t)i * NN + j] - mx);
  s = blockSum(s, red, tid);
  const float lse = mx + __logf(s);
  _Float16* orow = At + (size_t)j * NN;           // At row j holds column j of A
  for (int i = tid; i < NN; i += 256)
    orow[i] = (_Float16)__expf(la1[(size_t)i * NN + j] - lse);
}

// ---------- kernel 3: per-block W = sinkhorn(W_1[n] . W_V); store Wt[n][o][i] f16 ----------
__global__ __launch_bounds__(256) void kMakeW(const float* __restrict__ W1,
                                              const float* __restrict__ WV,
                                              _Float16* __restrict__ Wt) {
  __shared__ float tile[64 * 65];
  __shared__ float rowlse[64];
  const int n = blockIdx.x, tid = threadIdx.x;
  float w1[8];
  #pragma unroll
  for (int c = 0; c < 8; c++) w1[c] = W1[n * 8 + c];
  #pragma unroll
  for (int t = 0; t < 16; t++) {
    int idx = tid + t * 256;
    int d = idx >> 6, e = idx & 63;
    float s = 0.f;
    #pragma unroll
    for (int c = 0; c < 8; c++) s += w1[c] * WV[c * 4096 + d * 64 + e];
    tile[d * 65 + e] = s * TEMP_INV;
  }
  __syncthreads();
  if (tid < 64) {                                  // row LSE (over e)
    float mx = -1e30f;
    for (int e = 0; e < 64; e++) mx = fmaxf(mx, tile[tid * 65 + e]);
    float s = 0.f;
    for (int e = 0; e < 64; e++) s += __expf(tile[tid * 65 + e] - mx);
    rowlse[tid] = mx + __logf(s);
  }
  __syncthreads();
  if (tid < 64) {                                  // col LSE (over d) + emit
    const int e = tid;
    float mx = -1e30f;
    for (int d = 0; d < 64; d++) mx = fmaxf(mx, tile[d * 65 + e] - rowlse[d]);
    float s = 0.f;
    for (int d = 0; d < 64; d++) s += __expf(tile[d * 65 + e] - rowlse[d] - mx);
    const float lse = mx + __logf(s);
    _Float16* dst = Wt + (size_t)n * 4096 + (size_t)e * 64;   // transposed: [o][i]
    for (int d = 0; d < 64; d++)
      dst[d] = (_Float16)__expf(tile[d * 65 + e] - rowlse[d] - lse);
  }
}

// ---------- kernel 4: local GEMM x_local[b,n,:] = x[b,n,:] @ W[n]  (WMMA) ----------
__global__ __launch_bounds__(256) void kLocal(const float* __restrict__ x,
                                              const _Float16* __restrict__ Wt,
                                              float* __restrict__ xloc) {
  __shared__ _Float16 As[64 * 72];   // rows = b-local, cols = i
  __shared__ _Float16 Bs[64 * 72];   // rows = o, cols = i   (from Wt)
  const int tid  = threadIdx.x;
  const int lane = tid & 31;
  const int wid  = tid >> 5;
  const int lrow = lane & 15;
  const int lhi  = lane >> 4;
  const int b0   = blockIdx.x * 64;
  const int n    = blockIdx.y;
  const int waveM = (wid & 3) * 16;   // 4 waves along b
  const int waveN = (wid >> 2) * 32;  // 2 waves along o

  // stage A: x[b0+row, n, 0..63] f32 -> f16 LDS
  {
    const int row = tid >> 2, part = tid & 3;
    const float4* src = (const float4*)(x + (size_t)(b0 + row) * NK + (size_t)n * KK + part * 16);
    _Float16* dst = &As[row * 72 + part * 16];
    #pragma unroll
    for (int q = 0; q < 2; q++) {
      float4 f0 = src[2 * q], f1 = src[2 * q + 1];
      v8h h = { (_Float16)f0.x, (_Float16)f0.y, (_Float16)f0.z, (_Float16)f0.w,
                (_Float16)f1.x, (_Float16)f1.y, (_Float16)f1.z, (_Float16)f1.w };
      *(v8h*)(dst + 8 * q) = h;
    }
    // stage B: Wt[n] 64x64 f16
    const v8h* wsrc = (const v8h*)(Wt + (size_t)n * 4096 + row * 64 + part * 16);
    _Float16* wdst = &Bs[row * 72 + part * 16];
    *(v8h*)(wdst)     = wsrc[0];
    *(v8h*)(wdst + 8) = wsrc[1];
  }
  __syncthreads();

  v8f acc[2] = {};
  #pragma unroll
  for (int ks = 0; ks < 64; ks += 32) {
    const int kbase = ks + lhi * 8;
    v16h af = ldsFrag(&As[(waveM + lrow) * 72 + kbase]);
    #pragma unroll
    for (int tn = 0; tn < 2; tn++) {
      v16h bf = ldsFrag(&Bs[(waveN + tn * 16 + lrow) * 72 + kbase]);
      acc[tn] = __builtin_amdgcn_wmma_f32_16x16x32_f16(false, af, false, bf,
                                                       (short)0, acc[tn], false, false);
    }
  }

  // epilogue: x_local[b, n, k] f32
  #pragma unroll
  for (int tn = 0; tn < 2; tn++) {
    V8F u; u.v = acc[tn];
    const int kcol = waveN + tn * 16 + lrow;
    const int brow = b0 + waveM + lhi * 8;
    #pragma unroll
    for (int v = 0; v < 8; v++)
      xloc[(size_t)(brow + v) * NK + (size_t)n * KK + kcol] = u.f[v];
  }
}

// ---------- kernel 5: LayerNorm over n for each (b,k); write xln[(b*K+k)][n] f16 ----------
__global__ __launch_bounds__(256) void kLN(const float* __restrict__ xloc,
                                           const float* __restrict__ gamma,
                                           const float* __restrict__ beta,
                                           _Float16* __restrict__ xln) {
  __shared__ float red[256];
  const int row = blockIdx.x;              // b*64 + k
  const int b = row >> 6, k = row & 63;
  const int tid = threadIdx.x;
  const float* base = xloc + (size_t)b * NK + k;
  float v[4], s = 0.f, ss = 0.f;
  #pragma unroll
  for (int t = 0; t < 4; t++) {
    const int nn = tid + t * 256;
    float xv = base[(size_t)nn * KK];
    v[t] = xv; s += xv; ss += xv * xv;
  }
  s  = blockSum(s,  red, tid);
  ss = blockSum(ss, red, tid);
  const float mean = s * (1.0f / NN);
  const float var  = ss * (1.0f / NN) - mean * mean;
  const float rstd = rsqrtf(var + LN_EPS);
  _Float16* orow = xln + (size_t)row * NN;
  #pragma unroll
  for (int t = 0; t < 4; t++) {
    const int nn = tid + t * 256;
    orow[nn] = (_Float16)((v[t] - mean) * rstd * gamma[nn] + beta[nn]);
  }
}

// ---------- kernel 6: big GEMM  out[b,m,k] = sum_n xln[(b,k)][n] * At[m][n] ----------
#define BKH 64    // k halfs per stage
#define LDB 72    // padded LDS row stride in halfs (144B, 16B-aligned)

__global__ __launch_bounds__(256) void kBig(const _Float16* __restrict__ xln,
                                            const _Float16* __restrict__ At,
                                            float* __restrict__ out) {
#if HAVE_ASYNC_LDS
  __shared__ _Float16 As[2][128 * LDB];
  __shared__ _Float16 Bs[2][128 * LDB];
#else
  __shared__ _Float16 As[1][128 * LDB];
  __shared__ _Float16 Bs[1][128 * LDB];
#endif
  const int tid  = threadIdx.x;
  const int lane = tid & 31;
  const int wid  = tid >> 5;
  const int lrow = lane & 15;
  const int lhi  = lane >> 4;
  const int m0 = blockIdx.y * 128;       // M tile base ((b,k) rows)
  const int n0 = blockIdx.x * 128;       // output-column tile base
  const int waveM = (wid >> 2) * 64;     // 2 waves along M
  const int waveN = (wid & 3) * 32;      // 4 waves along output cols
  const int srow = tid >> 1;             // 0..127 : staging row
  const int scol = (tid & 1) * 32;       // staging col (halfs): 2 threads/row x 64B

  const _Float16* gA = xln + (size_t)(m0 + srow) * NN + scol;
  const _Float16* gB = At  + (size_t)(n0 + srow) * NN + scol;
  _Float16* lA0 = &As[0][srow * LDB + scol];
  _Float16* lB0 = &Bs[0][srow * LDB + scol];

  v8f acc[4][2] = {};

  auto compute = [&](const _Float16* as, const _Float16* bs) {
    #pragma unroll
    for (int ks = 0; ks < BKH; ks += 32) {
      const int kbase = ks + lhi * 8;
      v16h af[4], bf[2];
      #pragma unroll
      for (int tm = 0; tm < 4; tm++)
        af[tm] = ldsFrag(&as[(waveM + tm * 16 + lrow) * LDB + kbase]);
      #pragma unroll
      for (int tn = 0; tn < 2; tn++)
        bf[tn] = ldsFrag(&bs[(waveN + tn * 16 + lrow) * LDB + kbase]);
      #pragma unroll
      for (int tm = 0; tm < 4; tm++)
        #pragma unroll
        for (int tn = 0; tn < 2; tn++)
          acc[tm][tn] = __builtin_amdgcn_wmma_f32_16x16x32_f16(false, af[tm], false, bf[tn],
                                                               (short)0, acc[tm][tn], false, false);
    }
  };

#if HAVE_ASYNC_LDS
  _Float16* lA1 = &As[1][srow * LDB + scol];
  _Float16* lB1 = &Bs[1][srow * LDB + scol];
  auto stage = [&](int kb, _Float16* la, _Float16* lb) {
    #pragma unroll
    for (int q = 0; q < 4; q++) {
      __builtin_amdgcn_global_load_async_to_lds_b128(asGlobal(gA + kb + q * 8),
                                                     asLds(la + q * 8), 0, 0);
      __builtin_amdgcn_global_load_async_to_lds_b128(asGlobal(gB + kb + q * 8),
                                                     asLds(lb + q * 8), 0, 0);
    }
  };
  stage(0, lA0, lB0);                 // prologue: fill buffer 0
  int cur = 0;
  for (int kb = 0; kb < NN; kb += BKH) {
    WAIT_ASYNC();                     // our async tiles for buf[cur] have landed
    __syncthreads();                  // everyone's tiles visible; prev reads done
    if (kb + BKH < NN)
      stage(kb + BKH, cur ? lA0 : lA1, cur ? lB0 : lB1);   // prefetch next buffer
    compute(cur ? &As[1][0] : &As[0][0], cur ? &Bs[1][0] : &Bs[0][0]);
    cur ^= 1;
  }
#else
  for (int kb = 0; kb < NN; kb += BKH) {
    uint4 ra[4], rb[4];
    #pragma unroll
    for (int q = 0; q < 4; q++) {
      ra[q] = *(const uint4*)(gA + kb + q * 8);
      rb[q] = *(const uint4*)(gB + kb + q * 8);
    }
    __syncthreads();
    #pragma unroll
    for (int q = 0; q < 4; q++) {
      *(uint4*)(lA0 + q * 8) = ra[q];
      *(uint4*)(lB0 + q * 8) = rb[q];
    }
    __syncthreads();
    compute(&As[0][0], &Bs[0][0]);
  }
#endif

  // epilogue with output transpose: out[b*NK + m*64 + k]
  #pragma unroll
  for (int tm = 0; tm < 4; tm++) {
    const int rowb = m0 + waveM + tm * 16 + lhi * 8;   // (b,k) row, 8-aligned
    const int b  = rowb >> 6;
    const int kk = rowb & 63;
    float* pb = out + (size_t)b * NK + kk;
    #pragma unroll
    for (int tn = 0; tn < 2; tn++) {
      const int m = n0 + waveN + tn * 16 + lrow;
      float* p = pb + (size_t)m * KK;
      V8F u; u.v = acc[tm][tn];
      *(float4*)(p)     = make_float4(u.f[0], u.f[1], u.f[2], u.f[3]);
      *(float4*)(p + 4) = make_float4(u.f[4], u.f[5], u.f[6], u.f[7]);
    }
  }
}

// ---------- launcher ----------
extern "C" void kernel_launch(void* const* d_in, const int* in_sizes, int n_in,
                              void* d_out, int out_size, void* d_ws, size_t ws_size,
                              hipStream_t stream) {
  const float* x    = (const float*)d_in[0];
  const float* Alog = (const float*)d_in[1];
  const float* W1   = (const float*)d_in[2];
  const float* WV   = (const float*)d_in[3];
  const float* g2   = (const float*)d_in[4];
  const float* b2   = (const float*)d_in[5];
  float* out = (float*)d_out;

  char* ws = (char*)d_ws;
  float*    la1  = (float*)(ws);                      // 4 MiB  [N,N] f32
  _Float16* At   = (_Float16*)(ws + (4ull  << 20));   // 2 MiB  A^T f16
  _Float16* Wt   = (_Float16*)(ws + (6ull  << 20));   // 8 MiB  W^T per block f16
  float*    xloc = (float*)(ws + (16ull << 20));      // 64 MiB x_local f32
  _Float16* xln  = (_Float16*)(ws + (80ull << 20));   // 32 MiB layernormed f16

  kSinkARow<<<NN, 256, 0, stream>>>(Alog, la1);
  kSinkACol<<<NN, 256, 0, stream>>>(la1, At);
  kMakeW  <<<NN, 256, 0, stream>>>(W1, WV, Wt);
  kLocal  <<<dim3(BB / 64, NN), 256, 0, stream>>>(x, Wt, xloc);
  kLN     <<<BB * KK, 256, 0, stream>>>(xloc, g2, b2, xln);
  kBig    <<<dim3(NN / 128, (BB * KK) / 128), 256, 0, stream>>>(xln, At, out);
}
